// TorchAttractorLanguageModel_38027640438857
// MI455X (gfx1250) — matrix operand order repaired and checked by reading, not compiled, not run on hardware
//
#include <hip/hip_runtime.h>

// ---------------------------------------------------------------------------
// CDNA5 (gfx1250) attractor LM: wave32 + v_wmma_f32_16x16x32_bf16
// ---------------------------------------------------------------------------

typedef __attribute__((ext_vector_type(16))) __bf16 v16bf;
typedef __attribute__((ext_vector_type(8)))  __bf16 v8bf;
typedef __attribute__((ext_vector_type(8)))  float  v8f;

#define D_DIM   512
#define T_LEN   128
#define B_DIM   256
#define V_DIM   32000
#define ROWS    16          // batch rows per workgroup (== WMMA M)
#define RS      516         // padded fp32 LDS row stride (floats)
#define FBS     520         // padded bf16 LDS row stride (bf16 elems)
#define BLK1    512         // 16 wave32s
#define DT_C    0.04f
#define CUBIC_C 0.008f
#define LAMBDA_C 0.1f
#define SIGSCALE 0.5f
#define NORM_EPS 1e-8f
#define SIG_EPS  1e-6f
#define LN_EPS   1e-5f
#define W_SLOW   0.3f

// ---- WMMA helpers ---------------------------------------------------------

__device__ __forceinline__ v8f wmma_bf16(v16bf a, v16bf b, v8f c) {
  // D = A(16x32 bf16) * B(32x16 bf16) + C(16x16 f32)
  return __builtin_amdgcn_wmma_f32_16x16x32_bf16(
      /*neg_a=*/false, a, /*neg_b=*/false, b,
      /*c_mod=*/(short)0, c, /*reuse_a=*/false, /*reuse_b=*/false);
}

// A fragment (16x32 bf16) from a bf16 LDS tile, row stride `stride`.
// lanes 0-15: M=lane,    K = kb+0..15
// lanes16-31: M=lane-16, K = kb+16..31
// Two 16-byte ds_load_b128s per fragment, no conversion.
__device__ __forceinline__ v16bf load_a_bf16(const __bf16* base, int lane, int kb,
                                             int stride) {
  const int r    = lane & 15;
  const int koff = (lane >> 4) << 4;            // 0 or 16
  const __bf16* p = base + r * stride + kb + koff;
  v8bf lo = *(const v8bf*)p;                    // K = kb+koff+0..7
  v8bf hi = *(const v8bf*)(p + 8);              // K = kb+koff+8..15
  return __builtin_shufflevector(lo, hi, 0,1,2,3,4,5,6,7,8,9,10,11,12,13,14,15);
}

// B fragment (32x16 bf16) = W^T tile: B[K,N] = W[n0+N][kb+K], W row-major ld=512.
// lanes 0-15: N=lane,    K = kb+0..15 (2 per VGPR)
// lanes16-31: N=lane-16, K = kb+16..31
// Four global_load_b128s + packed converts.
__device__ __forceinline__ v16bf load_b_glb(const float* __restrict__ W,
                                            int lane, int n0, int kb) {
  const int col  = n0 + (lane & 15);
  const float4* p = (const float4*)(W + (size_t)col * D_DIM + kb + ((lane >> 4) << 4));
  const float4 q0 = p[0], q1 = p[1], q2 = p[2], q3 = p[3];
  v16bf v;
  v[0]=(__bf16)q0.x;  v[1]=(__bf16)q0.y;  v[2]=(__bf16)q0.z;  v[3]=(__bf16)q0.w;
  v[4]=(__bf16)q1.x;  v[5]=(__bf16)q1.y;  v[6]=(__bf16)q1.z;  v[7]=(__bf16)q1.w;
  v[8]=(__bf16)q2.x;  v[9]=(__bf16)q2.y;  v[10]=(__bf16)q2.z; v[11]=(__bf16)q2.w;
  v[12]=(__bf16)q3.x; v[13]=(__bf16)q3.y; v[14]=(__bf16)q3.z; v[15]=(__bf16)q3.w;
  return v;
}

// Store 16x16 f32 accumulator tile to LDS (row stride RS).
// VGPR p: lanes 0-15 -> (M=p, N=n0+lane); lanes 16-31 -> (M=p+8, N=n0+lane-16)
__device__ __forceinline__ void store_tile(float* mat, v8f c, int lane, int n0) {
  const int n  = n0 + (lane & 15);
  const int mb = (lane >> 4) << 3;
#pragma unroll
  for (int p = 0; p < 8; ++p) mat[(p + mb) * RS + n] = c[p];
}

// ---- per-row reduction: 32 threads per row, returns full row sum ----------
__device__ __forceinline__ float rowreduce(float* red, float v, int row, int chunk) {
  red[row * 33 + chunk] = v;
  __syncthreads();
  if (chunk == 0) {
    float s = 0.f;
#pragma unroll
    for (int i = 0; i < 32; ++i) s += red[row * 33 + i];
    red[row * 33 + 32] = s;
  }
  __syncthreads();
  return red[row * 33 + 32];
}

// ===========================================================================
// Kernel 1: full 128-token recurrence. 16 WGs x 512 thr; 16 rows per WG.
// ===========================================================================
__global__ __launch_bounds__(BLK1, 1)
void attractor_recurrence(const int*   __restrict__ tokens,
                          const float* __restrict__ embed,
                          const float* __restrict__ diffusion,
                          const float* __restrict__ gamma_p,
                          const float* __restrict__ beta_p,
                          float*       __restrict__ combined_out) {
  extern __shared__ float smem[];
  float* fastS = smem;                    // ROWS*RS fp32
  float* slowS = fastS + ROWS * RS;
  float* sigS  = slowS + ROWS * RS;
  float* matS  = sigS  + ROWS * RS;
  float* red   = matS  + ROWS * RS;       // ROWS*33
  __bf16* fastBF = (__bf16*)(red + ROWS * 33);  // ROWS*FBS bf16 mirror of fast

  const int tid   = threadIdx.x;
  const int lane  = tid & 31;
  const int wave  = tid >> 5;             // 0..15
  const int row   = tid >> 5;             // elementwise row == wave index
  const int chunk = tid & 31;             // 32 threads per row
  const int c0    = chunk * 16;           // 16 contiguous cols per thread
  const int j0    = wave * 32;            // this wave's 32 output features
  const int gB    = blockIdx.x * ROWS;    // global batch base

  const float gamma = *gamma_p;
  const float beta  = *beta_p;

  // zero persistent state
  for (int i = tid; i < ROWS * RS; i += BLK1) { fastS[i] = 0.f; slowS[i] = 0.f; }
  for (int i = tid; i < ROWS * FBS; i += BLK1) fastBF[i] = (__bf16)0.f;
  __syncthreads();

  // warm WGP$/L2 with this wave's diffusion columns (global_prefetch_b8)
  for (int jj = 0; jj < 32; ++jj)
    __builtin_prefetch(diffusion + (size_t)(j0 + jj) * D_DIM + (lane << 4), 0, 1);

  for (int t = 0; t < T_LEN; ++t) {
    // ---- token embed -> layernorm -> unit base direction -----------------
    const int tok = tokens[(size_t)(gB + row) * T_LEN + t];
    const float* erow = embed + (size_t)tok * D_DIM;
    float x[16];
    float s1 = 0.f, s2 = 0.f;
#pragma unroll
    for (int i = 0; i < 16; ++i) {
      x[i] = erow[c0 + i];
      s1 += x[i]; s2 += x[i] * x[i];
    }
    const float sum   = rowreduce(red, s1, row, chunk);
    const float sumsq = rowreduce(red, s2, row, chunk);
    const float mu  = sum * (1.0f / D_DIM);
    float var = sumsq * (1.0f / D_DIM) - mu * mu;
    var = fmaxf(var, 0.f);
    const float rstd  = rsqrtf(var + LN_EPS);
    const float lnorm = sqrtf((float)D_DIM * var / (var + LN_EPS)); // ||LN(x)||
    const float binv  = rstd / fmaxf(lnorm, 1e-12f);                // base = (x-mu)*binv

    // ---- context vector norms -------------------------------------------
    float fv[16], cv[16];
    float f2 = 0.f, cb2 = 0.f;
#pragma unroll
    for (int i = 0; i < 16; ++i) {
      fv[i] = fastS[row * RS + c0 + i];
      cv[i] = fv[i] + W_SLOW * slowS[row * RS + c0 + i];
      f2 += fv[i] * fv[i]; cb2 += cv[i] * cv[i];
    }
    const float fn = sqrtf(rowreduce(red, f2,  row, chunk));
    const float cn = sqrtf(rowreduce(red, cb2, row, chunk));

    // ---- signal = normalize(base + gamma*ctx) ---------------------------
    float sg2 = 0.f;
    float sv[16];
#pragma unroll
    for (int i = 0; i < 16; ++i) {
      float ctx = (fn > 1e-8f) ? fv[i] / (fn + SIG_EPS)
                : ((cn > 1e-8f) ? cv[i] / (cn + SIG_EPS) : 0.f);
      float s = (x[i] - mu) * binv + gamma * ctx;
      sv[i] = s; sg2 += s * s;
    }
    const float sn   = sqrtf(rowreduce(red, sg2, row, chunk));
    const float sinv = 1.f / (sn + SIG_EPS);
#pragma unroll
    for (int i = 0; i < 16; ++i) sigS[row * RS + c0 + i] = sv[i] * sinv;
    __syncthreads();

    // ---- 4 attractor convergence steps ----------------------------------
    for (int cs = 0; cs < 4; ++cs) {
      float psum = 0.f;
#pragma unroll
      for (int i = 0; i < 16; ++i) psum += fastS[row * RS + c0 + i];
      const float mu_f = rowreduce(red, psum, row, chunk) * (1.0f / D_DIM);

      // matS = fast @ diffusion^T   (wave: 2 N-tiles, K = 16 steps of 32)
      v8f acc0 = {0.f,0.f,0.f,0.f,0.f,0.f,0.f,0.f};
      v8f acc1 = {0.f,0.f,0.f,0.f,0.f,0.f,0.f,0.f};
#pragma unroll 2
      for (int k = 0; k < 16; ++k) {
        const int kb = k * 32;
        v16bf a  = load_a_bf16(fastBF, lane, kb, FBS);
        v16bf b0 = load_b_glb(diffusion, lane, j0,      kb);
        v16bf b1 = load_b_glb(diffusion, lane, j0 + 16, kb);
        acc0 = wmma_bf16(a, b0, acc0);
        acc1 = wmma_bf16(a, b1, acc1);
      }
      store_tile(matS, acc0, lane, j0);
      store_tile(matS, acc1, lane, j0 + 16);
      __syncthreads();

      // elementwise Euler step + nan_to_num + renormalize + clip
      float snew[16], n2 = 0.f;
#pragma unroll
      for (int i = 0; i < 16; ++i) {
        const float f = fastS[row * RS + c0 + i];
        const float drift = matS[row * RS + c0 + i]
                          + CUBIC_C * tanhf(f - mu_f)
                          + beta * (SIGSCALE * sigS[row * RS + c0 + i])
                          - LAMBDA_C * f;
        float s = f + DT_C * drift;
        if (!(fabsf(s) < 1e38f)) s = 0.f;            // NaN/Inf -> 0
        snew[i] = s; n2 += s * s;
      }
      const float nrm = sqrtf(rowreduce(red, n2, row, chunk));
      const float inv = 1.f / (nrm + NORM_EPS);
#pragma unroll
      for (int i = 0; i < 16; ++i) {
        float v = snew[i] * inv;
        v = fminf(fmaxf(v, -10.f), 10.f);
        fastS[row * RS + c0 + i]  = v;
        fastBF[row * FBS + c0 + i] = (__bf16)v;      // keep bf16 mirror in sync
      }
      __syncthreads();
    }

    // ---- slow state update ----------------------------------------------
    float slv[16], sl2 = 0.f;
#pragma unroll
    for (int i = 0; i < 16; ++i) {
      float s = 0.95f * slowS[row * RS + c0 + i] + 0.05f * fastS[row * RS + c0 + i];
      slv[i] = s; sl2 += s * s;
    }
    const float snrm  = sqrtf(rowreduce(red, sl2, row, chunk));
    const float scale = (snrm > 0.5f) ? 0.5f / (snrm + 1e-12f) : 1.f;
#pragma unroll
    for (int i = 0; i < 16; ++i) slowS[row * RS + c0 + i] = slv[i] * scale;
    __syncthreads();
  }

  // combined = fast + 0.3*slow -> workspace
#pragma unroll
  for (int i = 0; i < 16; ++i)
    combined_out[(size_t)(gB + row) * D_DIM + c0 + i] =
        fastS[row * RS + c0 + i] + W_SLOW * slowS[row * RS + c0 + i];
}

// ===========================================================================
// Kernel 2: logits[256,32000] = combined @ readout^T, WMMA bf16.
// 250 WGs x 256 thr: WG = 128 M-rows (staged bf16 in LDS) x 256 N-cols.
// ===========================================================================
#define MB_ROWS 128
#define AS      520   // padded bf16 LDS row stride

__global__ __launch_bounds__(256, 1)
void attractor_logits(const float* __restrict__ combined,
                      const float* __restrict__ readout,
                      float*       __restrict__ out) {
  extern __shared__ __bf16 aS[];          // MB_ROWS * AS
  const int tid   = threadIdx.x;
  const int lane  = tid & 31;
  const int wave  = tid >> 5;             // 0..7
  const int mblk  = blockIdx.x & 1;
  const int ngrp  = blockIdx.x >> 1;
  const int mbase = mblk * MB_ROWS;
  const int nbase = ngrp * 256 + wave * 32;

  // stage A slice as bf16
  for (int i = tid; i < MB_ROWS * D_DIM; i += 256) {
    const int r = i >> 9, c = i & 511;
    aS[r * AS + c] = (__bf16)combined[(size_t)(mbase + r) * D_DIM + c];
  }
  __syncthreads();

  v8f acc[8][2];
  const v8f zero = {0.f,0.f,0.f,0.f,0.f,0.f,0.f,0.f};
#pragma unroll
  for (int m = 0; m < 8; ++m) { acc[m][0] = zero; acc[m][1] = zero; }

#pragma unroll 1
  for (int k = 0; k < 16; ++k) {
    const int kb = k * 32;
    v16bf b0 = load_b_glb(readout, lane, nbase,      kb);
    v16bf b1 = load_b_glb(readout, lane, nbase + 16, kb);
#pragma unroll
    for (int m = 0; m < 8; ++m) {
      v16bf a = load_a_bf16(aS + m * 16 * AS, lane, kb, AS);
      acc[m][0] = wmma_bf16(a, b0, acc[m][0]);
      acc[m][1] = wmma_bf16(a, b1, acc[m][1]);
    }
  }

  const int nloc = lane & 15;
  const int madd = (lane >> 4) << 3;
#pragma unroll
  for (int m = 0; m < 8; ++m) {
#pragma unroll
    for (int p = 0; p < 8; ++p) {
      const int gm = mbase + m * 16 + p + madd;
      out[(size_t)gm * V_DIM + nbase + nloc]      = acc[m][0][p];
      out[(size_t)gm * V_DIM + nbase + 16 + nloc] = acc[m][1][p];
    }
  }
}

// ===========================================================================
extern "C" void kernel_launch(void* const* d_in, const int* in_sizes, int n_in,
                              void* d_out, int out_size, void* d_ws, size_t ws_size,
                              hipStream_t stream) {
  const int*   tokens    = (const int*)  d_in[0];   // [256,128]
  const float* embed     = (const float*)d_in[1];   // [32000,512]
  const float* readout   = (const float*)d_in[2];   // [32000,512]
  const float* diffusion = (const float*)d_in[3];   // [512,512]
  const float* gamma_p   = (const float*)d_in[4];
  const float* beta_p    = (const float*)d_in[5];
  float* out      = (float*)d_out;                  // [256,32000]
  float* combined = (float*)d_ws;                   // [256,512] scratch

  const size_t smem1 = (size_t)(4 * ROWS * RS + ROWS * 33) * sizeof(float)
                     + (size_t)ROWS * FBS * 2;      // + bf16 fast mirror (~147 KB)
  attractor_recurrence<<<B_DIM / ROWS, BLK1, smem1, stream>>>(
      tokens, embed, diffusion, gamma_p, beta_p, combined);

  const size_t smem2 = (size_t)MB_ROWS * AS * 2;    // bf16 stage, ~130 KB
  attractor_logits<<<(B_DIM / MB_ROWS) * (V_DIM / 256), 256, smem2, stream>>>(
      combined, readout, out);
}
